// Encoder_43705587204671
// MI455X (gfx1250) — matrix operand order repaired and checked by reading, not compiled, not run on hardware
//
#include <hip/hip_runtime.h>

// Problem constants (from reference)
#define BDIM 64     // batch
#define TDIM 512    // time steps
#define NE   1024   // hidden size
#define H4   4096   // 4*NE gate width
#define K2   2048   // 2*NE GEMM K
#define NBLK 256    // persistent grid: 64 hid tiles * 4 M tiles

typedef __attribute__((ext_vector_type(16))) _Float16 v16h;
typedef __attribute__((ext_vector_type(8)))  _Float16 v8h;
typedef __attribute__((ext_vector_type(8)))  float    v8f;

__device__ __forceinline__ float sigmoidf(float x) {
    return 1.0f / (1.0f + __expf(-x));
}
__device__ __forceinline__ float fast_tanhf(float x) {
    // tanh(x) = 2*sigmoid(2x) - 1
    return 2.0f / (1.0f + __expf(-2.0f * x)) - 1.0f;
}

// -------------------------------------------------------------------------
// Pack one layer's fp32 weight matrix W[2048,4096] (row-major) into f16 in
// exact WMMA B-fragment order so the GEMM does one contiguous 32B load/lane:
//   Wt[((n_tile*64 + kb)*32 + lane)*16 + i]
//     = W[(kb*32 + (i<8?0:16) + (lane<16?0:8) + (i&7)) * 4096 + n_tile*16 + (lane&15)]
// -------------------------------------------------------------------------
__global__ void k_pack_w(const float* __restrict__ W, _Float16* __restrict__ Wt) {
    int o = blockIdx.x * 256 + threadIdx.x;     // 0 .. 8388607
    int n_tile = o >> 15;                       // 64*512 elems per n-tile
    int r      = o & 32767;
    int kb     = r >> 9;
    int q      = r & 511;
    int lane   = q >> 4;
    int i      = q & 15;
    int k = kb * 32 + ((i < 8) ? 0 : 16) + ((lane < 16) ? 0 : 8) + (i & 7);
    int n = n_tile * 16 + (lane & 15);
    Wt[o] = (_Float16)W[(size_t)k * H4 + n];
}

// fp32 -> f16 elementwise (forward-direction layer-0 input)
__global__ void k_f32_to_f16(const float* __restrict__ src, _Float16* __restrict__ dst) {
    size_t o = (size_t)blockIdx.x * 256 + threadIdx.x;
    dst[o] = (_Float16)src[o];
}

// tf.reverse_sequence + f16 convert (backward-direction layer-0 input)
__global__ void k_reverse_f16(const float* __restrict__ e, const int* __restrict__ nlen,
                              _Float16* __restrict__ x) {
    size_t o = (size_t)blockIdx.x * 256 + threadIdx.x;
    int b = (int)(o >> 19);          // T*NE = 2^19
    int t = (int)((o >> 10) & 511);
    int k = (int)(o & 1023);
    int nb = nlen[b];
    int st = (t < nb) ? (nb - 1 - t) : t;
    x[o] = (_Float16)e[((size_t)b * TDIM + st) * NE + k];
}

__global__ void k_zero_state(_Float16* __restrict__ h, float* __restrict__ c,
                             unsigned* __restrict__ bar) {
    int i = blockIdx.x * 256 + threadIdx.x;   // 65536 threads
    h[i] = (_Float16)0.0f;
    c[i] = 0.0f;
    if (i == 0) *bar = 0u;
}

__global__ void k_h_to_f32(const _Float16* __restrict__ h, float* __restrict__ o) {
    int i = blockIdx.x * 256 + threadIdx.x;
    o[i] = (float)h[i];
}

// -------------------------------------------------------------------------
// Persistent LSTM layer kernel: runs ALL 512 time steps of one layer in a
// single launch.  grid = (64 hidden tiles, 4 M tiles), block = 32 (one wave
// per 16x16 tile).  Per step: g = [x_t, h] @ W + b (M=64,K=2048,N=4096);
// each wave owns one hidden tile and computes its i/j/f/o gate tiles, then
// the gate nonlinearities + length masking in registers.  Steps are
// separated by a device-wide barrier: a monotonically increasing agent-scope
// atomic counter (release on arrive, acquire + s_sleep poll on wait), with h
// ping-ponged between two buffers so step t+1 never overwrites data a
// straggler still needs.
// -------------------------------------------------------------------------
__global__ void __launch_bounds__(32)
k_lstm_seq(const _Float16* __restrict__ x, const _Float16* __restrict__ Wt,
           const float* __restrict__ bias, _Float16* __restrict__ ha,
           _Float16* __restrict__ hb, float* __restrict__ c,
           const int* __restrict__ nlen, _Float16* __restrict__ y,
           unsigned* __restrict__ bar)
{
    const int lane     = threadIdx.x;      // wave32
    const int hid_tile = blockIdx.x;       // 0..63
    const int m_tile   = blockIdx.y;       // 0..3

    // A-fragment row for this lane (16-bit A 16x32 layout: M = lane%16)
    const int mrow = m_tile * 16 + (lane & 15);
    const int koff = (lane >= 16) ? 8 : 0; // K half-group swizzle

    // Per-gate W fragment streams: gate g covers n_tile = g*64 + hid_tile;
    // each kb step is 512 contiguous halves, lane reads 16 halves (32B).
    const size_t tile_blk = 64 * 512;      // halves per n-tile
    const _Float16* wp0 = Wt + (size_t)(0 * 64 + hid_tile) * tile_blk + lane * 16;
    const _Float16* wp1 = Wt + (size_t)(1 * 64 + hid_tile) * tile_blk + lane * 16;
    const _Float16* wp2 = Wt + (size_t)(2 * 64 + hid_tile) * tile_blk + lane * 16;
    const _Float16* wp3 = Wt + (size_t)(3 * 64 + hid_tile) * tile_blk + lane * 16;

    // Warm the W stream into L2 (global_prefetch_b8); once per kernel.
    __builtin_prefetch(wp0, 0, 0);
    __builtin_prefetch(wp1, 0, 0);
    __builtin_prefetch(wp2, 0, 0);
    __builtin_prefetch(wp3, 0, 0);

    // Step-invariant pointwise operands, hoisted out of the time loop.
    const int hid = hid_tile * 16 + (lane & 15);
    const float bi = bias[0 * NE + hid];
    const float bj = bias[1 * NE + hid];
    const float bf = bias[2 * NE + hid];
    const float bo = bias[3 * NE + hid];
    int nl[8];
    #pragma unroll
    for (int v = 0; v < 8; ++v)
        nl[v] = nlen[m_tile * 16 + v + ((lane < 16) ? 0 : 8)];

    union AV { v16h v; v8h h2[2]; };

    for (int t = 0; t < TDIM; ++t) {
        const _Float16* h_in  = (t & 1) ? hb : ha;
        _Float16*       h_out = (t & 1) ? ha : hb;
        const _Float16* xrow = x    + ((size_t)mrow * TDIM + t) * NE;
        const _Float16* hrow = h_in + (size_t)mrow * NE;

        v8f acc0 = {}, acc1 = {}, acc2 = {}, acc3 = {};

        // Prologue: fragments for kb = 0 (A from x since k0 = 0 < NE)
        AV A;
        A.h2[0] = *(const v8h*)(xrow + koff);
        A.h2[1] = *(const v8h*)(xrow + 16 + koff);
        v16h B0 = *(const v16h*)wp0;
        v16h B1 = *(const v16h*)wp1;
        v16h B2 = *(const v16h*)wp2;
        v16h B3 = *(const v16h*)wp3;

        for (int kb = 0; kb < 64; ++kb) {
            // Next-iteration loads into a second register set so they
            // overlap the WMMAs below (last iteration reloads harmlessly).
            const int kn = (kb + 1 < 64) ? (kb + 1) : kb;
            const int k0 = kn * 32;
            // Both 8-half A segments sit on the same side of the x/h
            // boundary (NE is a multiple of 32): one base select per iter.
            const _Float16* abase = (k0 < NE) ? (xrow + k0) : (hrow + (k0 - NE));
            AV An;
            An.h2[0] = *(const v8h*)(abase + koff);
            An.h2[1] = *(const v8h*)(abase + 16 + koff);
            const size_t wo = (size_t)kn * 512;
            v16h Bn0 = *(const v16h*)(wp0 + wo);
            v16h Bn1 = *(const v16h*)(wp1 + wo);
            v16h Bn2 = *(const v16h*)(wp2 + wo);
            v16h Bn3 = *(const v16h*)(wp3 + wo);

            acc0 = __builtin_amdgcn_wmma_f32_16x16x32_f16(false, A.v, false, B0, (short)0, acc0, false, false);
            acc1 = __builtin_amdgcn_wmma_f32_16x16x32_f16(false, A.v, false, B1, (short)0, acc1, false, false);
            acc2 = __builtin_amdgcn_wmma_f32_16x16x32_f16(false, A.v, false, B2, (short)0, acc2, false, false);
            acc3 = __builtin_amdgcn_wmma_f32_16x16x32_f16(false, A.v, false, B3, (short)0, acc3, false, false);

            A = An; B0 = Bn0; B1 = Bn1; B2 = Bn2; B3 = Bn3;
        }

        // Gate pointwise. C/D layout: N = lane%16, M = v + (lane<16 ? 0 : 8)
        #pragma unroll
        for (int v = 0; v < 8; ++v) {
            const int m = m_tile * 16 + v + ((lane < 16) ? 0 : 8);
            const size_t sidx = (size_t)m * NE + hid;
            float gi = acc0[v] + bi;
            float gj = acc1[v] + bj;
            float gf = acc2[v] + bf;
            float go = acc3[v] + bo;
            float c_old = c[sidx];
            float c_new = c_old * sigmoidf(gf + 1.0f) + sigmoidf(gi) * fast_tanhf(gj);
            float h_new = fast_tanhf(c_new) * sigmoidf(go);
            bool  valid = (t < nl[v]);
            float h_old = (float)h_in[sidx];
            c[sidx]     = valid ? c_new : c_old;
            h_out[sidx] = (_Float16)(valid ? h_new : h_old);
            y[((size_t)m * TDIM + t) * NE + hid] = (_Float16)(valid ? h_new : 0.0f);
        }

        // ---- device-wide step barrier (monotonic counter, no reset) ----
        __threadfence();   // make h_out/c stores visible before arriving
        if (lane == 0) {
            __hip_atomic_fetch_add(bar, 1u, __ATOMIC_ACQ_REL,
                                   __HIP_MEMORY_SCOPE_AGENT);
            const unsigned target = (unsigned)(t + 1) * (unsigned)NBLK;
            while (__hip_atomic_load(bar, __ATOMIC_ACQUIRE,
                                     __HIP_MEMORY_SCOPE_AGENT) < target) {
                __builtin_amdgcn_s_sleep(2);
            }
        }
        // lanes reconverge; lane0's acquire (with device-scope invalidate)
        // ordered the wave's subsequent loads of the other blocks' h.
    }
}

// -------------------------------------------------------------------------
// s = [hT_fw, hT_bw] @ P ; j = mean((s - s_)^2).  128 outputs, tiny.
// -------------------------------------------------------------------------
__global__ void __launch_bounds__(128)
k_final(const float* __restrict__ hTfw, const float* __restrict__ hTbw,
        const float* __restrict__ P, const float* __restrict__ s_tgt,
        float* __restrict__ out)
{
    __shared__ float red[128];
    const int tid = threadIdx.x;       // 0..127
    const int b = tid >> 1, cc = tid & 1;
    float s = 0.0f;
    for (int k = 0; k < NE; ++k) {
        s += hTfw[b * NE + k] * P[k * 2 + cc];
        s += hTbw[b * NE + k] * P[(NE + k) * 2 + cc];
    }
    out[tid] = s;
    float d = s - s_tgt[tid];
    red[tid] = d * d;
    __syncthreads();
    for (int off = 64; off > 0; off >>= 1) {
        if (tid < off) red[tid] += red[tid + off];
        __syncthreads();
    }
    if (tid == 0) out[128] = red[0] / 128.0f;
}

// -------------------------------------------------------------------------
extern "C" void kernel_launch(void* const* d_in, const int* in_sizes, int n_in,
                              void* d_out, int out_size, void* d_ws, size_t ws_size,
                              hipStream_t stream)
{
    (void)in_sizes; (void)n_in; (void)out_size; (void)ws_size;
    const float* e     = (const float*)d_in[0];
    const int*   nlen  = (const int*)d_in[1];
    const float* s_tgt = (const float*)d_in[2];
    const float* W_fw  = (const float*)d_in[3];
    const float* b_fw  = (const float*)d_in[4];
    const float* W_bw  = (const float*)d_in[5];
    const float* b_bw  = (const float*)d_in[6];
    const float* P     = (const float*)d_in[7];
    float* out = (float*)d_out;

    // Workspace carve (~152 MB): packed W (16.8MB), two f16 activation
    // buffers (64MB each), state + final-h scratch + barrier counter.
    char* w = (char*)d_ws;
    _Float16* Wt  = (_Float16*)w;  w += (size_t)K2 * H4 * 2;
    _Float16* xb  = (_Float16*)w;  w += (size_t)BDIM * TDIM * NE * 2;
    _Float16* yb  = (_Float16*)w;  w += (size_t)BDIM * TDIM * NE * 2;
    _Float16* h0  = (_Float16*)w;  w += (size_t)BDIM * NE * 2;
    _Float16* h1  = (_Float16*)w;  w += (size_t)BDIM * NE * 2;
    float*    cb  = (float*)w;     w += (size_t)BDIM * NE * 4;
    float*    hTf = (float*)w;     w += (size_t)BDIM * NE * 4;
    float*    hTb = (float*)w;     w += (size_t)BDIM * NE * 4;
    unsigned* bar = (unsigned*)w;  w += 256;

    const size_t NW = (size_t)K2 * H4;          // 8,388,608 weight elems/layer
    const size_t NX = (size_t)BDIM * TDIM * NE; // 33,554,432 activation elems
    dim3 seqGrid(64, 4), seqBlk(32);

    for (int dir = 0; dir < 2; ++dir) {
        const float* W    = dir ? W_bw : W_fw;
        const float* bias = dir ? b_bw : b_fw;
        float*       hT   = dir ? hTb  : hTf;

        if (dir == 0)
            k_f32_to_f16<<<(unsigned)(NX / 256), 256, 0, stream>>>(e, xb);
        else
            k_reverse_f16<<<(unsigned)(NX / 256), 256, 0, stream>>>(e, nlen, xb);

        for (int l = 0; l < 2; ++l) {
            const _Float16* xin  = (l == 0) ? xb : yb;
            _Float16*       yout = (l == 0) ? yb : xb;  // layer-1 y is scratch
            k_pack_w<<<(unsigned)(NW / 256), 256, 0, stream>>>(W + (size_t)l * NW, Wt);
            k_zero_state<<<(BDIM * NE) / 256, 256, 0, stream>>>(h0, cb, bar);
            // One persistent launch runs all 512 steps of this layer.
            k_lstm_seq<<<seqGrid, seqBlk, 0, stream>>>(
                xin, Wt, bias + (size_t)l * H4, h0, h1, cb, nlen, yout, bar);
            if (l == 1)  // T even -> final h landed in h0 (ha)
                k_h_to_f32<<<(BDIM * NE) / 256, 256, 0, stream>>>(h0, hT);
        }
    }
    k_final<<<1, 128, 0, stream>>>(hTf, hTb, P, s_tgt, out);
}
